// AttentionModel_3444563771775
// MI455X (gfx1250) — matrix operand order repaired and checked by reading, not compiled, not run on hardware
//
#include <hip/hip_runtime.h>
#include <hip/hip_bf16.h>

typedef __attribute__((ext_vector_type(16))) _Float16 v16h;
typedef __attribute__((ext_vector_type(8)))  float    v8f;

#define BATCH 256
#define NNODE 1000
#define DIM   128
#define CTXD  130
#define NHEAD 8

// One workgroup per batch row b. 256 threads = 8 waves (wave32).
__global__ __launch_bounds__(256)
void am_decoder_kernel(const float* __restrict__ emb,      // [B, N, D]
                       const float* __restrict__ sctx,     // [B, 1, CTX]
                       const unsigned char* __restrict__ mask, // [B, 1, N] bool
                       const float* __restrict__ Wn,       // [D, 3D]
                       const float* __restrict__ Wf,       // [D, D]
                       const float* __restrict__ Ws,       // [CTX, D]
                       const float* __restrict__ Wo,       // [D, D]
                       float* __restrict__ out)            // [B, 1, N]
{
    const int b    = blockIdx.x;
    const int t    = threadIdx.x;
    const int wave = t >> 5;
    const int lane = t & 31;
    const float NEG_INF = -__builtin_inff();

    __shared__ float    wlds[NNODE * NHEAD];   // compat -> attn weights, 32000 B
    __shared__ _Float16 qkt[16 * DIM];         // QK^T, f16, head rows 8..15 zero
    __shared__ float    slds[NHEAD * DIM];     // s_h vectors
    __shared__ float    llds[NNODE];           // final logits
    __shared__ float    meanv[DIM], qv[DIM], hv[DIM], gl[DIM], g2v[DIM];
    __shared__ float    scs[132];
    __shared__ float    red[256];
    __shared__ float    mh[NHEAD], zh[NHEAD];

    const float* eb      = emb + (size_t)b * NNODE * DIM;
    const unsigned char* mrow = mask + (size_t)b * NNODE;

    // ---- Phase 0: graph mean over nodes (and stage step_context) ----
    if (t < CTXD) scs[t] = sctx[(size_t)b * CTXD + t];
    {
        const int d = t & 127, half = t >> 7;
        float acc = 0.f;
        for (int n = half; n < NNODE; n += 2) acc += eb[n * DIM + d];
        red[t] = acc;
    }
    __syncthreads();
    if (t < DIM) meanv[t] = (red[t] + red[t + 128]) * (1.0f / (float)NNODE);
    __syncthreads();

    // ---- Phase 1: query = mean@Wf + sc@Ws ----
    if (t < DIM) {
        float acc = 0.f;
        for (int k = 0; k < DIM; ++k)  acc += meanv[k] * Wf[k * DIM + t];
        for (int c = 0; c < CTXD; ++c) acc += scs[c]   * Ws[c * DIM + t];
        qv[t] = acc;
    }
    __syncthreads();

    // ---- Phase 2: QK^T[h][k] = 0.25 * sum_j Wk[k][h*16+j] * q[h*16+j] ----
    for (int idx = t; idx < DIM * NHEAD; idx += 256) {
        const int k = idx >> 3, h = idx & 7;
        float acc = 0.f;
        for (int j = 0; j < 16; ++j) acc += Wn[k * 384 + h * 16 + j] * qv[h * 16 + j];
        qkt[h * DIM + k] = (_Float16)(acc * 0.25f);     // fold 1/sqrt(hd)
    }
    for (int idx = t; idx < DIM * NHEAD; idx += 256)    // pad head rows 8..15
        qkt[NHEAD * DIM + idx] = (_Float16)0.0f;
    __syncthreads();

    // ---- Phase 3: compat^T = QK^T x E-chunks via v_wmma_f32_16x16x32_f16 ----
    {
        // A fragments: 16x32 f16 lane layout (lane: M=lane&15; K pairs per VGPR)
        v16h afrag[4];
        const int M  = lane & 15;
        const int kb = (lane < 16) ? 0 : 8;
        #pragma unroll
        for (int ks = 0; ks < 4; ++ks) {
            #pragma unroll
            for (int v = 0; v < 8; ++v) {
                const int kk = ks * 32 + kb + ((v < 4) ? (2 * v) : (16 + 2 * (v - 4)));
                afrag[ks][2 * v]     = qkt[M * DIM + kk];
                afrag[ks][2 * v + 1] = qkt[M * DIM + kk + 1];
            }
        }
        for (int tile = wave; tile < 63; tile += 8) {
            const int n0 = tile * 16;
            const int r  = n0 + (lane & 15);            // node row for this lane
            const float* erow = eb + (size_t)r * DIM;
            v8f c = {0.f, 0.f, 0.f, 0.f, 0.f, 0.f, 0.f, 0.f};
            #pragma unroll
            for (int ks = 0; ks < 4; ++ks) {
                // B fragment 32x16 f16: lane n-col = lane&15, K in table order
                const int kbase = ks * 32 + ((lane < 16) ? 0 : 8);
                v16h bfrag;
                if (r < NNODE) {
                    const float4* p = reinterpret_cast<const float4*>(erow + kbase);
                    #pragma unroll
                    for (int q4 = 0; q4 < 4; ++q4) {
                        const float4 f = p[q4 + ((q4 < 2) ? 0 : 2)];
                        bfrag[q4 * 4 + 0] = (_Float16)f.x;
                        bfrag[q4 * 4 + 1] = (_Float16)f.y;
                        bfrag[q4 * 4 + 2] = (_Float16)f.z;
                        bfrag[q4 * 4 + 3] = (_Float16)f.w;
                    }
                } else {
                    #pragma unroll
                    for (int i = 0; i < 16; ++i) bfrag[i] = (_Float16)0.0f;
                }
                c = __builtin_amdgcn_wmma_f32_16x16x32_f16(
                        false, afrag[ks], false, bfrag, (short)0, c, false, false);
            }
            // C layout: lanes 0..15 hold M=j (=head j) for node r in VGPR j
            if (lane < 16 && r < NNODE) {
                const bool msk = (mrow[r] != 0);
                #pragma unroll
                for (int j = 0; j < NHEAD; ++j)
                    wlds[r * NHEAD + j] = msk ? NEG_INF : c[j];
            }
        }
    }
    __syncthreads();

    // ---- Phase 4: per-head masked softmax stats; wlds := exp(compat - m_h) ----
    {
        const int h = wave;
        float m = NEG_INF;
        for (int n = lane; n < NNODE; n += 32) m = fmaxf(m, wlds[n * NHEAD + h]);
        for (int off = 16; off > 0; off >>= 1) m = fmaxf(m, __shfl_xor(m, off, 32));
        float zs = 0.f;
        for (int n = lane; n < NNODE; n += 32) {
            const float e = expf(wlds[n * NHEAD + h] - m);
            wlds[n * NHEAD + h] = e;
            zs += e;
        }
        for (int off = 16; off > 0; off >>= 1) zs += __shfl_xor(zs, off, 32);
        if (lane == 0) { mh[h] = m; zh[h] = zs; }
    }
    __syncthreads();

    // ---- Phase 5: s_h = sum_n attn_hn * e_n (4 heads per thread, d = t&127) ----
    {
        const int d  = t & 127;
        const int hg = (t >> 7) * 4;                    // head group 0..3 or 4..7
        float a0 = 0.f, a1 = 0.f, a2 = 0.f, a3 = 0.f;
        #pragma unroll 4
        for (int n = 0; n < NNODE; ++n) {
            const float e = eb[n * DIM + d];
            const float* w = &wlds[n * NHEAD + hg];     // LDS broadcast in-wave
            a0 += e * w[0]; a1 += e * w[1]; a2 += e * w[2]; a3 += e * w[3];
        }
        slds[(hg + 0) * DIM + d] = a0;
        slds[(hg + 1) * DIM + d] = a1;
        slds[(hg + 2) * DIM + d] = a2;
        slds[(hg + 3) * DIM + d] = a3;
    }
    __syncthreads();

    // ---- Phase 6: hvec[d] = (s_h @ Wv[:,d]) / z_h  (h = d/16) ----
    if (t < DIM) {
        const int h = t >> 4;
        float acc = 0.f;
        for (int k = 0; k < DIM; ++k) acc += slds[h * DIM + k] * Wn[k * 384 + 128 + t];
        hv[t] = acc / zh[h];
    }
    __syncthreads();

    // ---- Phase 7: glimpse = hvec @ W_out ----
    if (t < DIM) {
        float acc = 0.f;
        for (int d = 0; d < DIM; ++d) acc += hv[d] * Wo[d * DIM + t];
        gl[t] = acc;
    }
    __syncthreads();

    // ---- Phase 8: g2[k] = (Wl[k,:]·glimpse) / sqrt(D) ----
    if (t < DIM) {
        float acc = 0.f;
        for (int d = 0; d < DIM; ++d) acc += gl[d] * Wn[t * 384 + 256 + d];
        g2v[t] = acc * 0.088388347648318447f;           // 1/sqrt(128)
    }
    __syncthreads();

    // ---- Phase 9: logits_n = tanh(e_n · g2)*10, masked (one wave per node) ----
    for (int n = wave; n < NNODE; n += 8) {
        const float4 f = reinterpret_cast<const float4*>(eb + (size_t)n * DIM)[lane];
        float p = f.x * g2v[lane * 4 + 0] + f.y * g2v[lane * 4 + 1]
                + f.z * g2v[lane * 4 + 2] + f.w * g2v[lane * 4 + 3];
        for (int off = 16; off > 0; off >>= 1) p += __shfl_xor(p, off, 32);
        if (lane == 0)
            llds[n] = mrow[n] ? NEG_INF : (tanhf(p) * 10.0f);
    }
    __syncthreads();

    // ---- Phase 10: log_softmax over nodes ----
    float m = NEG_INF;
    for (int n = t; n < NNODE; n += 256) m = fmaxf(m, llds[n]);
    red[t] = m;
    __syncthreads();
    for (int s = 128; s > 0; s >>= 1) {
        if (t < s) red[t] = fmaxf(red[t], red[t + s]);
        __syncthreads();
    }
    const float mm = red[0];
    __syncthreads();
    float zs = 0.f;
    for (int n = t; n < NNODE; n += 256) zs += expf(llds[n] - mm);
    red[t] = zs;
    __syncthreads();
    for (int s = 128; s > 0; s >>= 1) {
        if (t < s) red[t] += red[t + s];
        __syncthreads();
    }
    const float logz = logf(red[0]) + mm;
    for (int n = t; n < NNODE; n += 256)
        out[(size_t)b * NNODE + n] = llds[n] - logz;
}

extern "C" void kernel_launch(void* const* d_in, const int* in_sizes, int n_in,
                              void* d_out, int out_size, void* d_ws, size_t ws_size,
                              hipStream_t stream) {
    (void)in_sizes; (void)n_in; (void)out_size; (void)d_ws; (void)ws_size;
    const float*         emb  = (const float*)d_in[0];
    const float*         sctx = (const float*)d_in[1];
    const unsigned char* mask = (const unsigned char*)d_in[2];  // jax bool = 1 byte
    const float*         Wn   = (const float*)d_in[3];
    const float*         Wf   = (const float*)d_in[4];
    const float*         Ws   = (const float*)d_in[5];
    const float*         Wo   = (const float*)d_in[6];
    float*               out  = (float*)d_out;
    am_decoder_kernel<<<BATCH, 256, 0, stream>>>(emb, sctx, mask, Wn, Wf, Ws, Wo, out);
}
